// OutData_46248207843549
// MI455X (gfx1250) — compile-verified, matrix operand order checked by reading, not run on hardware
//
#include <hip/hip_runtime.h>
#include <hip/hip_bf16.h>
#include <stdint.h>

typedef __attribute__((ext_vector_type(16))) _Float16 v16h;
typedef __attribute__((ext_vector_type(8)))  _Float16 v8h;
typedef __attribute__((ext_vector_type(8)))  float    v8f;

union V16u { v16h v; v8h h[2]; };

__device__ __forceinline__ float silu_f(float x) {
    return __fdividef(x, 1.f + __expf(-x));
}

__device__ __forceinline__ v8f wmma_f16(v16h a, v16h b, v8f c) {
    return __builtin_amdgcn_wmma_f32_16x16x32_f16(false, a, false, b, (short)0, c, false, false);
}

// Build a 16x32 f16 A/B fragment (wave32 layout): lane%16 = row (M or N),
// K halves = kb..kb+7 and kb+16..kb+23, kb = kc*32 + (lane>=16)*8.
__device__ __forceinline__ v16h frag_h16(const _Float16* p, int kb) {
    V16u u;
    u.h[0] = *(const v8h*)(p + kb);
    u.h[1] = *(const v8h*)(p + kb + 16);
    return u.v;
}

// Same fragment but converting from f32 source on the fly.
__device__ __forceinline__ v16h frag_f32(const float* p, int kb) {
    V16u u;
#pragma unroll
    for (int i = 0; i < 8; ++i) {
        u.v[i]     = (_Float16)p[kb + i];
        u.v[i + 8] = (_Float16)p[kb + 16 + i];
    }
    return u.v;
}

// CDNA5 async global->LDS copy: 16B per lane, tracked by ASYNCcnt.
// ldsaddr = LDS byte address (generic-pointer low 32 bits), gaddr = per-lane global addr.
__device__ __forceinline__ void async_ld_b128(uint32_t ldsaddr, const float* gaddr) {
    asm volatile("global_load_async_to_lds_b128 %0, %1, off"
                 :: "v"(ldsaddr), "v"(gaddr)
                 : "memory");
}

__device__ __forceinline__ void wait_asynccnt0() {
    asm volatile("s_wait_asynccnt 0x0" ::: "memory");
}

__global__ void cvt_f16_kernel(const float* __restrict__ x, _Float16* __restrict__ y, int n) {
    int i = blockIdx.x * blockDim.x + threadIdx.x;
    if (i < n) y[i] = (_Float16)x[i];
}

// ---------------- Energy path: silu MLP 128->64->32->16, dot with Woute ----------------
__global__ __launch_bounds__(256)
void energy_kernel(const _Float16* __restrict__ hA,
                   const _Float16* __restrict__ W0,   // 64x128 f16
                   const _Float16* __restrict__ W1,   // 32x64  f16
                   const _Float16* __restrict__ W2,   // 16x32  f16
                   const float* __restrict__ Wout,    // 16 f32
                   const float* __restrict__ bout,    // 1 f32
                   float* __restrict__ out, int N)
{
    __shared__ _Float16 smem[8 * 16 * 64];
    const int lane = threadIdx.x & 31;
    const int wv   = threadIdx.x >> 5;
    const int l16  = lane & 15;
    const int hi   = lane >> 4;
    const int ntiles = (N + 15) >> 4;
    const int tile = blockIdx.x * 8 + wv;
    const int t    = tile < ntiles ? tile : (ntiles - 1);
    const int row0 = t * 16;
    _Float16* buf = smem + wv * (16 * 64);

    const int arow = min(row0 + l16, N - 1);
    v16h a[4];
#pragma unroll
    for (int kc = 0; kc < 4; ++kc)
        a[kc] = frag_h16(hA + (size_t)arow * 128, kc * 32 + hi * 8);

    // layer0: 128 -> 64
#pragma unroll
    for (int nt = 0; nt < 4; ++nt) {
        v8f acc = {0.f,0.f,0.f,0.f,0.f,0.f,0.f,0.f};
#pragma unroll
        for (int kc = 0; kc < 4; ++kc)
            acc = wmma_f16(a[kc], frag_h16(W0 + (size_t)(nt * 16 + l16) * 128, kc * 32 + hi * 8), acc);
#pragma unroll
        for (int r = 0; r < 8; ++r)
            buf[(r + 8 * hi) * 64 + nt * 16 + l16] = (_Float16)silu_f(acc[r]);
    }
    __syncthreads();

    v16h a1[2];
#pragma unroll
    for (int kc = 0; kc < 2; ++kc)
        a1[kc] = frag_h16(buf + l16 * 64, kc * 32 + hi * 8);
    __syncthreads();

    // layer1: 64 -> 32
#pragma unroll
    for (int nt = 0; nt < 2; ++nt) {
        v8f acc = {0.f,0.f,0.f,0.f,0.f,0.f,0.f,0.f};
#pragma unroll
        for (int kc = 0; kc < 2; ++kc)
            acc = wmma_f16(a1[kc], frag_h16(W1 + (size_t)(nt * 16 + l16) * 64, kc * 32 + hi * 8), acc);
#pragma unroll
        for (int r = 0; r < 8; ++r)
            buf[(r + 8 * hi) * 32 + nt * 16 + l16] = (_Float16)silu_f(acc[r]);
    }
    __syncthreads();

    v16h a2 = frag_h16(buf + l16 * 32, hi * 8);
    __syncthreads();

    // layer2: 32 -> 16
    v8f acc = {0.f,0.f,0.f,0.f,0.f,0.f,0.f,0.f};
    acc = wmma_f16(a2, frag_h16(W2 + (size_t)l16 * 32, hi * 8), acc);

    const float w  = Wout[l16];
    const float bo = bout[0];
#pragma unroll
    for (int r = 0; r < 8; ++r) {
        float s = silu_f(acc[r]) * w;
        s += __shfl_xor(s, 1, 32);
        s += __shfl_xor(s, 2, 32);
        s += __shfl_xor(s, 4, 32);
        s += __shfl_xor(s, 8, 32);
        const int m = row0 + r + 8 * hi;
        if (l16 == 0 && tile < ntiles && m < N) out[m] = s + bo;
    }
}

// ------- Edge path: rbf@Wrbf.T, *edge, silu MLP 128->64->32->16, dot Woutf -> raw -------
// 4 waves / block: LDS = 4 * (8KB f32 edge stage + 4KB f16 activations) = 48KB
__global__ __launch_bounds__(128)
void edge_kernel(const float* __restrict__ edge,
                 const float* __restrict__ rbf,     // Ex64 f32
                 const _Float16* __restrict__ Wr,   // 128x64 f16
                 const _Float16* __restrict__ W0,   // 64x128 f16
                 const _Float16* __restrict__ W1,   // 32x64  f16
                 const _Float16* __restrict__ W2,   // 16x32  f16
                 const float* __restrict__ Wout,    // 16 f32
                 float* __restrict__ raw, int E)
{
    __shared__ float    eshm[4 * 16 * 128];   // f32 edge tiles (async staged)
    __shared__ _Float16 smem[4 * 16 * 128];   // f16 activation tiles
    const int lane = threadIdx.x & 31;
    const int wv   = threadIdx.x >> 5;
    const int l16  = lane & 15;
    const int hi   = lane >> 4;
    const int ntiles = (E + 15) >> 4;
    const int tile = blockIdx.x * 4 + wv;
    const int t    = tile < ntiles ? tile : (ntiles - 1);
    const int row0 = t * 16;
    _Float16* buf  = smem + wv * (16 * 128);
    float*    ebuf = eshm + wv * (16 * 128);

    // Kick off async staging of the 16x128 f32 edge tile (one 512B row per op),
    // overlapping with the rbf fragment conversion below.
    {
        const uint32_t lbase = (uint32_t)(uintptr_t)ebuf + (uint32_t)(lane * 16);
#pragma unroll
        for (int j = 0; j < 16; ++j) {
            const int rr = min(row0 + j, E - 1);
            async_ld_b128(lbase + (uint32_t)(j * 512), edge + (size_t)rr * 128 + lane * 4);
        }
    }

    const int erow = min(row0 + l16, E - 1);
    v16h ra[2];
#pragma unroll
    for (int kc = 0; kc < 2; ++kc)
        ra[kc] = frag_f32(rbf + (size_t)erow * 64, kc * 32 + hi * 8);

    wait_asynccnt0();

    // rbf_f = rbf @ Wrbf.T (64 -> 128), fuse elementwise edge multiply, stage f16 in LDS
#pragma unroll
    for (int nt = 0; nt < 8; ++nt) {
        v8f acc = {0.f,0.f,0.f,0.f,0.f,0.f,0.f,0.f};
#pragma unroll
        for (int kc = 0; kc < 2; ++kc)
            acc = wmma_f16(ra[kc], frag_h16(Wr + (size_t)(nt * 16 + l16) * 64, kc * 32 + hi * 8), acc);
#pragma unroll
        for (int r = 0; r < 8; ++r) {
            const float ev = ebuf[(r + 8 * hi) * 128 + nt * 16 + l16];
            buf[(r + 8 * hi) * 128 + nt * 16 + l16] = (_Float16)(ev * acc[r]);
        }
    }
    __syncthreads();

    v16h a[4];
#pragma unroll
    for (int kc = 0; kc < 4; ++kc)
        a[kc] = frag_h16(buf + l16 * 128, kc * 32 + hi * 8);
    __syncthreads();

    // layer0: 128 -> 64
#pragma unroll
    for (int nt = 0; nt < 4; ++nt) {
        v8f acc = {0.f,0.f,0.f,0.f,0.f,0.f,0.f,0.f};
#pragma unroll
        for (int kc = 0; kc < 4; ++kc)
            acc = wmma_f16(a[kc], frag_h16(W0 + (size_t)(nt * 16 + l16) * 128, kc * 32 + hi * 8), acc);
#pragma unroll
        for (int r = 0; r < 8; ++r)
            buf[(r + 8 * hi) * 64 + nt * 16 + l16] = (_Float16)silu_f(acc[r]);
    }
    __syncthreads();

    v16h a1[2];
#pragma unroll
    for (int kc = 0; kc < 2; ++kc)
        a1[kc] = frag_h16(buf + l16 * 64, kc * 32 + hi * 8);
    __syncthreads();

    // layer1: 64 -> 32
#pragma unroll
    for (int nt = 0; nt < 2; ++nt) {
        v8f acc = {0.f,0.f,0.f,0.f,0.f,0.f,0.f,0.f};
#pragma unroll
        for (int kc = 0; kc < 2; ++kc)
            acc = wmma_f16(a1[kc], frag_h16(W1 + (size_t)(nt * 16 + l16) * 64, kc * 32 + hi * 8), acc);
#pragma unroll
        for (int r = 0; r < 8; ++r)
            buf[(r + 8 * hi) * 32 + nt * 16 + l16] = (_Float16)silu_f(acc[r]);
    }
    __syncthreads();

    v16h a2 = frag_h16(buf + l16 * 32, hi * 8);
    __syncthreads();

    // layer2: 32 -> 16
    v8f acc = {0.f,0.f,0.f,0.f,0.f,0.f,0.f,0.f};
    acc = wmma_f16(a2, frag_h16(W2 + (size_t)l16 * 32, hi * 8), acc);

    const float w = Wout[l16];
#pragma unroll
    for (int r = 0; r < 8; ++r) {
        float s = silu_f(acc[r]) * w;
        s += __shfl_xor(s, 1, 32);
        s += __shfl_xor(s, 2, 32);
        s += __shfl_xor(s, 4, 32);
        s += __shfl_xor(s, 8, 32);
        const int m = row0 + r + 8 * hi;
        if (l16 == 0 && tile < ntiles && m < E) raw[m] = s;
    }
}

// ---------------- cosine similarity of gathered atom rows, one wave / edge ----------------
__global__ __launch_bounds__(256)
void eps_kernel(const float* __restrict__ atom, const int* __restrict__ nbr,
                float* __restrict__ eps, int E)
{
    const int lane = threadIdx.x & 31;
    const int wv   = threadIdx.x >> 5;
    const int e = blockIdx.x * 8 + wv;
    if (e >= E) return;
    const int i0 = nbr[2 * e + 0];
    const int i1 = nbr[2 * e + 1];
    const float4* p0 = (const float4*)(atom + (size_t)i0 * 128);
    const float4* p1 = (const float4*)(atom + (size_t)i1 * 128);
    float4 x = p0[lane];
    float4 y = p1[lane];
    float d  = x.x*y.x + x.y*y.y + x.z*y.z + x.w*y.w;
    float n0 = x.x*x.x + x.y*x.y + x.z*x.z + x.w*x.w;
    float n1 = y.x*y.x + y.y*y.y + y.z*y.z + y.w*y.w;
#pragma unroll
    for (int m = 16; m >= 1; m >>= 1) {
        d  += __shfl_xor(d,  m, 32);
        n0 += __shfl_xor(n0, m, 32);
        n1 += __shfl_xor(n1, m, 32);
    }
    if (lane == 0) eps[e] = d * rsqrtf(n0 * n1);
}

// force = 0.5*(f + f[swap]),  f = -raw*eps
__global__ void combine_kernel(const float* __restrict__ raw, const float* __restrict__ eps,
                               const int* __restrict__ swp, float* __restrict__ f, int E)
{
    int i = blockIdx.x * blockDim.x + threadIdx.x;
    if (i >= E) return;
    int s = swp[i];
    f[i] = -0.5f * (raw[i] * eps[i] + raw[s] * eps[s]);
}

extern "C" void kernel_launch(void* const* d_in, const int* in_sizes, int n_in,
                              void* d_out, int out_size, void* d_ws, size_t ws_size,
                              hipStream_t stream) {
    const float* atom  = (const float*)d_in[0];
    const float* edge  = (const float*)d_in[1];
    const float* rbf   = (const float*)d_in[2];
    const int*   nbr   = (const int*)d_in[3];
    const int*   swp   = (const int*)d_in[4];
    const float* We0   = (const float*)d_in[5];
    const float* We1   = (const float*)d_in[6];
    const float* We2   = (const float*)d_in[7];
    const float* Woute = (const float*)d_in[8];
    const float* boute = (const float*)d_in[9];
    const float* Wf0   = (const float*)d_in[10];
    const float* Wf1   = (const float*)d_in[11];
    const float* Wf2   = (const float*)d_in[12];
    const float* Woutf = (const float*)d_in[13];
    const float* Wrbf  = (const float*)d_in[14];

    const int N = in_sizes[0] / 128;
    const int E = in_sizes[1] / 128;

    char* ws = (char*)d_ws;
    size_t off = 0;
    auto alloc = [&](size_t bytes) {
        char* p = ws + off;
        off = (off + bytes + 255) & ~(size_t)255;
        return p;
    };
    _Float16* hA   = (_Float16*)alloc((size_t)N * 128 * 2);
    _Float16* hWe0 = (_Float16*)alloc(64 * 128 * 2);
    _Float16* hWe1 = (_Float16*)alloc(32 * 64 * 2);
    _Float16* hWe2 = (_Float16*)alloc(16 * 32 * 2);
    _Float16* hWf0 = (_Float16*)alloc(64 * 128 * 2);
    _Float16* hWf1 = (_Float16*)alloc(32 * 64 * 2);
    _Float16* hWf2 = (_Float16*)alloc(16 * 32 * 2);
    _Float16* hWr  = (_Float16*)alloc(128 * 64 * 2);
    float* raw = (float*)alloc((size_t)E * 4);
    float* eps = (float*)alloc((size_t)E * 4);

    float* outE = (float*)d_out;
    float* outF = outE + N;

    const int nA = N * 128;
    cvt_f16_kernel<<<(nA + 255) / 256, 256, 0, stream>>>(atom, hA, nA);
    cvt_f16_kernel<<<32, 256, 0, stream>>>(We0, hWe0, 64 * 128);
    cvt_f16_kernel<<<8, 256, 0, stream>>>(We1, hWe1, 32 * 64);
    cvt_f16_kernel<<<2, 256, 0, stream>>>(We2, hWe2, 16 * 32);
    cvt_f16_kernel<<<32, 256, 0, stream>>>(Wf0, hWf0, 64 * 128);
    cvt_f16_kernel<<<8, 256, 0, stream>>>(Wf1, hWf1, 32 * 64);
    cvt_f16_kernel<<<2, 256, 0, stream>>>(Wf2, hWf2, 16 * 32);
    cvt_f16_kernel<<<32, 256, 0, stream>>>(Wrbf, hWr, 128 * 64);

    const int atiles = (N + 15) / 16;
    energy_kernel<<<(atiles + 7) / 8, 256, 0, stream>>>(hA, hWe0, hWe1, hWe2, Woute, boute, outE, N);

    const int etiles = (E + 15) / 16;
    edge_kernel<<<(etiles + 3) / 4, 128, 0, stream>>>(edge, rbf, hWr, hWf0, hWf1, hWf2, Woutf, raw, E);

    eps_kernel<<<(E + 7) / 8, 256, 0, stream>>>(atom, nbr, eps, E);
    combine_kernel<<<(E + 255) / 256, 256, 0, stream>>>(raw, eps, swp, outF, E);
}
